// AXK1MoE_7060926235238
// MI455X (gfx1250) — matrix operand order repaired and checked by reading, not compiled, not run on hardware
//
#include <hip/hip_runtime.h>
#include <stdint.h>

#define T_TOK   1024
#define HDIM    1024
#define IDIM    512
#define EDIM    16
#define TOPK    4
#define NGROUP  4
#define GSIZE   4      // EDIM / NGROUP
#define ISH     1024
#define RSCALE  2.5f
#define NROWS   (T_TOK * TOPK)   // 4096 compacted routed rows

typedef __attribute__((ext_vector_type(8)))  float  v8f;
typedef __attribute__((ext_vector_type(16))) __bf16 v16bf;

union FragB { uint4 q[2]; v16bf v; };

static __device__ inline unsigned short f2bf(float f) {
    unsigned u = __float_as_uint(f);
    unsigned r = u + 0x7FFFu + ((u >> 16) & 1u);
    return (unsigned short)(r >> 16);
}
static __device__ inline unsigned pack2bf(float a, float b) {
    return (unsigned)f2bf(a) | ((unsigned)f2bf(b) << 16);
}
static __device__ inline float fast_sigmoid(float x) {
    return __builtin_amdgcn_rcpf(1.0f + __expf(-x));
}
static __device__ inline float fast_silu(float x) {
    return x * __builtin_amdgcn_rcpf(1.0f + __expf(-x));
}

// ---- CDNA5 async global->LDS (per-lane), tracked by ASYNCcnt ----
static __device__ inline void async_ld_b128(unsigned lds_off_bytes, const void* gaddr) {
    asm volatile("global_load_async_to_lds_b128 %0, %1, off"
                 :: "v"(lds_off_bytes), "v"(gaddr) : "memory");
}
template <int N>
static __device__ inline void wait_async_le() {
    asm volatile("s_wait_asynccnt %0" :: "i"(N) : "memory");
}

// Load one lane's WMMA fragment (8 dwords, contiguous in LDS fragment order)
static __device__ inline v16bf ld_frag(const unsigned* base, int lane) {
    FragB f;
    const uint4* p = (const uint4*)base;
    f.q[0] = p[lane * 2];
    f.q[1] = p[lane * 2 + 1];
    return f.v;
}

// ---------------------------------------------------------------- small kernels
__global__ void moe_zero_kernel(int* cnt) {
    if (threadIdx.x < EDIM) cnt[threadIdx.x] = 0;
}

__global__ void moe_cvt_kernel(const float* __restrict__ x, unsigned short* __restrict__ xbf, int n2) {
    int i = blockIdx.x * blockDim.x + threadIdx.x;
    if (i < n2) ((unsigned*)xbf)[i] = pack2bf(x[2 * i], x[2 * i + 1]);
}

__global__ void moe_prefix_kernel(const int* __restrict__ cnt, int* __restrict__ offs) {
    if (threadIdx.x == 0) {
        int o = 0;
        for (int e = 0; e < EDIM; ++e) { offs[e] = o; o += cnt[e]; }
        offs[EDIM] = o;
    }
}

// ---- weight pre-pass: fp32 [K][N] -> bf16 transposed [N][K], LDS-tiled (both sides coalesced)
__global__ __launch_bounds__(256) void moe_transcvt_kernel(
    const float* __restrict__ src, unsigned short* __restrict__ dst,
    int K, int N, long long srcStride, long long dstStride)
{
    __shared__ float tile[64][65];
    int z = blockIdx.z;
    const float* S = src + (long long)z * srcStride;
    unsigned short* D = dst + (long long)z * dstStride;
    int k0 = blockIdx.y * 64, n0 = blockIdx.x * 64;
    int tid = threadIdx.x;
#pragma unroll
    for (int r = 0; r < 16; ++r) {
        int d = tid + (r << 8);
        int kl = d >> 6, nl = d & 63;
        tile[kl][nl] = S[(long long)(k0 + kl) * N + (n0 + nl)];
    }
    __syncthreads();
#pragma unroll
    for (int r = 0; r < 8; ++r) {
        int d = tid + (r << 8);
        int nl = d >> 5, kd = d & 31;
        unsigned v = pack2bf(tile[2 * kd][nl], tile[2 * kd + 1][nl]);
        ((unsigned*)D)[((long long)(n0 + nl) * K + k0 + 2 * kd) >> 1] = v;
    }
}

// ---------------------------------------------------------------- router
__global__ __launch_bounds__(128) void moe_router_kernel(
    const float* __restrict__ x, const float* __restrict__ rw, const float* __restrict__ ebias,
    int* __restrict__ cnt, int* __restrict__ eix, int* __restrict__ slt, float* __restrict__ wts)
{
    int wave = threadIdx.x >> 5, lane = threadIdx.x & 31;
    int t = blockIdx.x * 4 + wave;

    float acc[EDIM];
#pragma unroll
    for (int e = 0; e < EDIM; ++e) acc[e] = 0.0f;

    const float* xr = x + (long long)t * HDIM;
    for (int h = lane; h < HDIM; h += 32) {
        float xv = xr[h];
        const float* r = rw + (long long)h * EDIM;
#pragma unroll
        for (int e = 0; e < EDIM; ++e) acc[e] += xv * r[e];
    }
#pragma unroll
    for (int e = 0; e < EDIM; ++e)
        for (int o = 16; o; o >>= 1) acc[e] += __shfl_xor(acc[e], o, 32);

    if (lane == 0) {
        float s[EDIM], sc[EDIM];
#pragma unroll
        for (int e = 0; e < EDIM; ++e) { s[e] = fast_sigmoid(acc[e]); sc[e] = s[e] + ebias[e]; }

        float gsc[NGROUP];
#pragma unroll
        for (int g = 0; g < NGROUP; ++g) {
            float m1 = -1e30f, m2 = -1e30f;
#pragma unroll
            for (int i = 0; i < GSIZE; ++i) {
                float v = sc[g * GSIZE + i];
                if (v > m1) { m2 = m1; m1 = v; } else if (v > m2) { m2 = v; }
            }
            gsc[g] = m1 + m2;
        }
        int g1 = 0; float b1 = -1e30f;
        for (int g = 0; g < NGROUP; ++g) if (gsc[g] > b1) { b1 = gsc[g]; g1 = g; }
        int g2 = (g1 == 0) ? 1 : 0; float b2 = -1e30f;
        for (int g = 0; g < NGROUP; ++g) if (g != g1 && gsc[g] > b2) { b2 = gsc[g]; g2 = g; }

        unsigned elig = 0;
        for (int i = 0; i < GSIZE; ++i) { elig |= 1u << (g1 * GSIZE + i); elig |= 1u << (g2 * GSIZE + i); }

        int pick[TOPK]; unsigned taken = 0;
        for (int k = 0; k < TOPK; ++k) {
            int best = 0; float bv = -1e30f;
            for (int e = 0; e < EDIM; ++e) {
                if ((taken >> e) & 1u) continue;
                float v = ((elig >> e) & 1u) ? sc[e] : 0.0f;   // match masked top_k
                if (v > bv) { bv = v; best = e; }
            }
            pick[k] = best; taken |= 1u << best;
        }
        float wk[TOPK], wsum = 0.0f;
        for (int k = 0; k < TOPK; ++k) { wk[k] = s[pick[k]]; wsum += wk[k]; }
        float inv = 1.0f / (wsum + 1e-20f);
        for (int k = 0; k < TOPK; ++k) {
            int e = pick[k];
            int sl = atomicAdd(&cnt[e], 1);
            eix[t * TOPK + k] = e;
            slt[t * TOPK + k] = sl;
            wts[t * TOPK + k] = wk[k] * inv;
        }
    }
}

// ---------------------------------------------------------------- gather X rows into expert-compact layout
__global__ __launch_bounds__(128) void moe_gather_kernel(
    const unsigned short* __restrict__ xbf, unsigned short* __restrict__ xg,
    const int* __restrict__ eix, const int* __restrict__ slt, const int* __restrict__ offs)
{
    int t = blockIdx.x;
    const uint4* src = (const uint4*)(xbf + (long long)t * HDIM);
    uint4 val = src[threadIdx.x];
#pragma unroll
    for (int k = 0; k < TOPK; ++k) {
        int row = offs[eix[t * TOPK + k]] + slt[t * TOPK + k];
        uint4* dst = (uint4*)(xg + (long long)row * HDIM);
        dst[threadIdx.x] = val;
    }
}

// ---------------------------------------------------------------- fused gate/up GEMM + SiLU*up
// 256 thr (8 waves). Tile M=64 x N=64, K-step 32, double-buffered async b128 staging.
// A row-major bf16 [rows][K]; B transposed bf16 [N][K]; out bf16.
__global__ __launch_bounds__(256) void moe_gateup_kernel(
    const unsigned short* __restrict__ A,
    const unsigned short* __restrict__ BgT,
    const unsigned short* __restrict__ BuT,
    unsigned short* __restrict__ Hout,
    const int* __restrict__ cnt, const int* __restrict__ offs,
    int Kdim, int Ntot, long long wstride, int nrowsTotal, int NeShared)
{
    __shared__ unsigned Asm[2][1024];
    __shared__ unsigned Bgsm[2][1024];
    __shared__ unsigned Busm[2][1024];

    int z = blockIdx.z;
    int Ne, base;
    if (cnt) { Ne = cnt[z]; base = offs[z]; } else { Ne = NeShared; base = 0; }
    int m_off = blockIdx.y * 64;
    if (m_off >= Ne) return;
    int n_off = blockIdx.x * 64;

    int tid = threadIdx.x;
    int lane = tid & 31, wave = tid >> 5;
    int msub = wave >> 1, npair = wave & 1;

    // staging decomposition: each thread copies one 16B chunk per tile per K-step
    int sub  = tid >> 6;         // fragment
    int ln   = (tid >> 1) & 31;  // lane within fragment
    int half = tid & 1;          // low/high 4 dwords

    int am   = (sub << 4) + (ln & 15);
    int arow = base + m_off + am; if (arow >= nrowsTotal) arow = nrowsTotal - 1;
    int akb  = ((ln >> 4) << 3) + (half << 4);          // A K-offset of 8 contiguous bf16
    const unsigned short* pA = A + (long long)arow * Kdim + akb;

    int bn  = n_off + (sub << 4) + (ln & 15);
    int bkb = ((ln >> 4) << 3) + (half << 4);           // B K-offset of 8 contiguous bf16
    const unsigned short* pBg = BgT + (long long)z * wstride + (long long)bn * Kdim + bkb;
    const unsigned short* pBu = BuT + (long long)z * wstride + (long long)bn * Kdim + bkb;

    unsigned ldsA  = (unsigned)(uintptr_t)(&Asm[0][0])  + tid * 16;
    unsigned ldsBg = (unsigned)(uintptr_t)(&Bgsm[0][0]) + tid * 16;
    unsigned ldsBu = (unsigned)(uintptr_t)(&Busm[0][0]) + tid * 16;

    v8f accG[2] = {};
    v8f accU[2] = {};

    auto compute_step = [&](int b) {
        v16bf a = ld_frag(&Asm[b][msub << 8], lane);
#pragma unroll
        for (int t2 = 0; t2 < 2; ++t2) {
            int ns = (npair << 1) + t2;
            v16bf bg = ld_frag(&Bgsm[b][ns << 8], lane);
            v16bf bu = ld_frag(&Busm[b][ns << 8], lane);
            accG[t2] = __builtin_amdgcn_wmma_f32_16x16x32_bf16(false, a, false, bg, (short)0, accG[t2], false, false);
            accU[t2] = __builtin_amdgcn_wmma_f32_16x16x32_bf16(false, a, false, bu, (short)0, accU[t2], false, false);
        }
    };

    // prologue: stage K-step 0 into buffer 0
    async_ld_b128(ldsA,  pA);  pA  += 32;
    async_ld_b128(ldsBg, pBg); pBg += 32;
    async_ld_b128(ldsBu, pBu); pBu += 32;

    int nIter = Kdim >> 5;
    // steady state: branch-free issue-next -> drain-current -> compute
    for (int it = 0; it < nIter - 1; ++it) {
        int b = it & 1;
        unsigned o = (unsigned)(b ^ 1) * 4096u;
        async_ld_b128(ldsA + o,  pA);  pA  += 32;
        async_ld_b128(ldsBg + o, pBg); pBg += 32;
        async_ld_b128(ldsBu + o, pBu); pBu += 32;
        wait_async_le<3>();               // in-order: <=3 left => current buffer landed
        __syncthreads();
        compute_step(b);
        __syncthreads();
    }
    wait_async_le<0>();
    __syncthreads();
    compute_step((nIter - 1) & 1);

#pragma unroll
    for (int t2 = 0; t2 < 2; ++t2) {
        v8f g = accG[t2], u = accU[t2];
        int n = n_off + (((npair << 1) + t2) << 4) + (lane & 15);
#pragma unroll
        for (int j = 0; j < 8; ++j) {
            int m = m_off + (msub << 4) + ((lane >> 4) << 3) + j;
            if (m < Ne)
                Hout[(long long)(base + m) * Ntot + n] = f2bf(fast_silu(g[j]) * u[j]);
        }
    }
}

// ---------------------------------------------------------------- down GEMM (bf16 A, bf16 B^T -> f32 out)
__global__ __launch_bounds__(256) void moe_down_kernel(
    const unsigned short* __restrict__ A,
    const unsigned short* __restrict__ BT,
    float* __restrict__ Out,
    const int* __restrict__ cnt, const int* __restrict__ offs,
    int Kdim, int Ntot, long long wstride, int nrowsTotal, int NeShared)
{
    __shared__ unsigned Asm[2][1024];
    __shared__ unsigned Bsm[2][1024];

    int z = blockIdx.z;
    int Ne, base;
    if (cnt) { Ne = cnt[z]; base = offs[z]; } else { Ne = NeShared; base = 0; }
    int m_off = blockIdx.y * 64;
    if (m_off >= Ne) return;
    int n_off = blockIdx.x * 64;

    int tid = threadIdx.x;
    int lane = tid & 31, wave = tid >> 5;
    int msub = wave >> 1, npair = wave & 1;

    int sub  = tid >> 6;
    int ln   = (tid >> 1) & 31;
    int half = tid & 1;

    int am   = (sub << 4) + (ln & 15);
    int arow = base + m_off + am; if (arow >= nrowsTotal) arow = nrowsTotal - 1;
    int akb  = ((ln >> 4) << 3) + (half << 4);
    const unsigned short* pA = A + (long long)arow * Kdim + akb;

    int bn  = n_off + (sub << 4) + (ln & 15);
    int bkb = ((ln >> 4) << 3) + (half << 4);
    const unsigned short* pB = BT + (long long)z * wstride + (long long)bn * Kdim + bkb;

    unsigned ldsA = (unsigned)(uintptr_t)(&Asm[0][0]) + tid * 16;
    unsigned ldsB = (unsigned)(uintptr_t)(&Bsm[0][0]) + tid * 16;

    v8f acc[2] = {};

    auto compute_step = [&](int b) {
        v16bf a = ld_frag(&Asm[b][msub << 8], lane);
#pragma unroll
        for (int t2 = 0; t2 < 2; ++t2) {
            int ns = (npair << 1) + t2;
            v16bf bb = ld_frag(&Bsm[b][ns << 8], lane);
            acc[t2] = __builtin_amdgcn_wmma_f32_16x16x32_bf16(false, a, false, bb, (short)0, acc[t2], false, false);
        }
    };

    async_ld_b128(ldsA, pA); pA += 32;
    async_ld_b128(ldsB, pB); pB += 32;

    int nIter = Kdim >> 5;
    for (int it = 0; it < nIter - 1; ++it) {
        int b = it & 1;
        unsigned o = (unsigned)(b ^ 1) * 4096u;
        async_ld_b128(ldsA + o, pA); pA += 32;
        async_ld_b128(ldsB + o, pB); pB += 32;
        wait_async_le<2>();
        __syncthreads();
        compute_step(b);
        __syncthreads();
    }
    wait_async_le<0>();
    __syncthreads();
    compute_step((nIter - 1) & 1);

#pragma unroll
    for (int t2 = 0; t2 < 2; ++t2) {
        v8f c = acc[t2];
        int n = n_off + (((npair << 1) + t2) << 4) + (lane & 15);
#pragma unroll
        for (int j = 0; j < 8; ++j) {
            int m = m_off + (msub << 4) + ((lane >> 4) << 3) + j;
            if (m < Ne)
                Out[(long long)(base + m) * Ntot + n] = c[j];
        }
    }
}

// ---------------------------------------------------------------- combine: out = shared + 2.5 * sum_k w_k * oe[row_k]
__global__ __launch_bounds__(256) void moe_combine_kernel(
    float* __restrict__ out, const float* __restrict__ oe,
    const int* __restrict__ eix, const int* __restrict__ slt,
    const int* __restrict__ offs, const float* __restrict__ wts)
{
    int t = blockIdx.x;
    int r0 = offs[eix[t * TOPK + 0]] + slt[t * TOPK + 0];
    int r1 = offs[eix[t * TOPK + 1]] + slt[t * TOPK + 1];
    int r2 = offs[eix[t * TOPK + 2]] + slt[t * TOPK + 2];
    int r3 = offs[eix[t * TOPK + 3]] + slt[t * TOPK + 3];
    float w0 = wts[t * TOPK + 0], w1 = wts[t * TOPK + 1];
    float w2 = wts[t * TOPK + 2], w3 = wts[t * TOPK + 3];

    for (int h = threadIdx.x; h < HDIM; h += 256) {
        float r = w0 * oe[(long long)r0 * HDIM + h] + w1 * oe[(long long)r1 * HDIM + h] +
                  w2 * oe[(long long)r2 * HDIM + h] + w3 * oe[(long long)r3 * HDIM + h];
        out[(long long)t * HDIM + h] += RSCALE * r;
    }
}

// ---------------------------------------------------------------- launcher
extern "C" void kernel_launch(void* const* d_in, const int* in_sizes, int n_in,
                              void* d_out, int out_size, void* d_ws, size_t ws_size,
                              hipStream_t stream) {
    (void)in_sizes; (void)n_in; (void)out_size; (void)ws_size;
    const float* x  = (const float*)d_in[0];
    const float* rw = (const float*)d_in[1];
    const float* eb = (const float*)d_in[2];
    const float* wg = (const float*)d_in[3];
    const float* wu = (const float*)d_in[4];
    const float* wd = (const float*)d_in[5];
    const float* sg = (const float*)d_in[6];
    const float* su = (const float*)d_in[7];
    const float* sd = (const float*)d_in[8];
    float* out = (float*)d_out;

    char* ws = (char*)d_ws;
    size_t off = 0;
    auto alloc = [&](size_t b) { size_t r = off; off += (b + 255) & ~(size_t)255; return r; };
    int*   cnt  = (int*)(ws + alloc(EDIM * 4));
    int*   offs = (int*)(ws + alloc((EDIM + 1) * 4));
    int*   eix  = (int*)(ws + alloc((size_t)T_TOK * TOPK * 4));
    int*   slt  = (int*)(ws + alloc((size_t)T_TOK * TOPK * 4));
    float* wts  = (float*)(ws + alloc((size_t)T_TOK * TOPK * 4));
    unsigned short* xbf  = (unsigned short*)(ws + alloc((size_t)T_TOK * HDIM * 2));
    unsigned short* xg   = (unsigned short*)(ws + alloc((size_t)NROWS * HDIM * 2));
    unsigned short* hbuf = (unsigned short*)(ws + alloc((size_t)NROWS * IDIM * 2));
    float*          oeb  = (float*)(ws + alloc((size_t)NROWS * HDIM * 4));
    unsigned short* hsh  = (unsigned short*)(ws + alloc((size_t)T_TOK * ISH * 2));
    unsigned short* wgT  = (unsigned short*)(ws + alloc((size_t)EDIM * HDIM * IDIM * 2));
    unsigned short* wuT  = (unsigned short*)(ws + alloc((size_t)EDIM * HDIM * IDIM * 2));
    unsigned short* wdT  = (unsigned short*)(ws + alloc((size_t)EDIM * IDIM * HDIM * 2));
    unsigned short* sgT  = (unsigned short*)(ws + alloc((size_t)HDIM * ISH * 2));
    unsigned short* suT  = (unsigned short*)(ws + alloc((size_t)HDIM * ISH * 2));
    unsigned short* sdT  = (unsigned short*)(ws + alloc((size_t)ISH * HDIM * 2));

    // weight pre-pass: fp32 [K][N] -> bf16 [N][K]
    moe_transcvt_kernel<<<dim3(IDIM / 64, HDIM / 64, EDIM), 256, 0, stream>>>(
        wg, wgT, HDIM, IDIM, (long long)HDIM * IDIM, (long long)HDIM * IDIM);
    moe_transcvt_kernel<<<dim3(IDIM / 64, HDIM / 64, EDIM), 256, 0, stream>>>(
        wu, wuT, HDIM, IDIM, (long long)HDIM * IDIM, (long long)HDIM * IDIM);
    moe_transcvt_kernel<<<dim3(HDIM / 64, IDIM / 64, EDIM), 256, 0, stream>>>(
        wd, wdT, IDIM, HDIM, (long long)IDIM * HDIM, (long long)IDIM * HDIM);
    moe_transcvt_kernel<<<dim3(ISH / 64, HDIM / 64, 1), 256, 0, stream>>>(
        sg, sgT, HDIM, ISH, 0, 0);
    moe_transcvt_kernel<<<dim3(ISH / 64, HDIM / 64, 1), 256, 0, stream>>>(
        su, suT, HDIM, ISH, 0, 0);
    moe_transcvt_kernel<<<dim3(HDIM / 64, ISH / 64, 1), 256, 0, stream>>>(
        sd, sdT, ISH, HDIM, 0, 0);

    moe_zero_kernel<<<1, 32, 0, stream>>>(cnt);
    moe_cvt_kernel<<<(T_TOK * HDIM / 2) / 256, 256, 0, stream>>>(x, xbf, T_TOK * HDIM / 2);
    moe_router_kernel<<<T_TOK / 4, 128, 0, stream>>>(x, rw, eb, cnt, eix, slt, wts);
    moe_prefix_kernel<<<1, 32, 0, stream>>>(cnt, offs);
    moe_gather_kernel<<<T_TOK, 128, 0, stream>>>(xbf, xg, eix, slt, offs);

    // routed experts
    moe_gateup_kernel<<<dim3(IDIM / 64, T_TOK / 64, EDIM), 256, 0, stream>>>(
        xg, wgT, wuT, hbuf, cnt, offs, HDIM, IDIM, (long long)HDIM * IDIM, NROWS, 0);
    moe_down_kernel<<<dim3(HDIM / 64, T_TOK / 64, EDIM), 256, 0, stream>>>(
        hbuf, wdT, oeb, cnt, offs, IDIM, HDIM, (long long)IDIM * HDIM, NROWS, 0);

    // shared experts (single "expert": Ne=T, base=0)
    moe_gateup_kernel<<<dim3(ISH / 64, T_TOK / 64, 1), 256, 0, stream>>>(
        xbf, sgT, suT, hsh, nullptr, nullptr, HDIM, ISH, 0, T_TOK, T_TOK);
    moe_down_kernel<<<dim3(HDIM / 64, T_TOK / 64, 1), 256, 0, stream>>>(
        hsh, sdT, out, nullptr, nullptr, ISH, HDIM, 0, T_TOK, T_TOK);

    moe_combine_kernel<<<T_TOK, 256, 0, stream>>>(out, oeb, eix, slt, offs, wts);
}
